// DetectionTargetLayer_48627619725487
// MI455X (gfx1250) — compile-verified, matrix operand order checked by reading, not run on hardware
//
#include <hip/hip_runtime.h>
#include <hip/hip_bf16.h>
#include <math.h>

#define BB      16
#define NPROP   2000
#define MAXGT   100
#define HH      160
#define WWD     160
#define TRAIN   200
#define NUMPOS  66
#define NUMNEG  134
#define MH      28
#define MWD     28
#define SORTN   2048

// d_out layout (floats, concatenated in reference return order)
#define OFF_ROIS  0          // B*200*4 = 12800
#define OFF_CLS   12800      // B*200   = 3200
#define OFF_DELT  16000      // B*200*4 = 12800
#define OFF_MASK  28800      // B*200*28*28 = 2508800

typedef _Float16 h8   __attribute__((ext_vector_type(8)));
typedef _Float16 h16  __attribute__((ext_vector_type(16)));
typedef float    v8f_ __attribute__((ext_vector_type(8)));

// deterministic stateless hash -> (0,1]; stands in for threefry sampling
__device__ __forceinline__ float rnd01(unsigned s, unsigned i) {
  unsigned x = (s + 0x51edu) * 0x9E3779B9u ^ (i + 1u) * 0x85EBCA6Bu;
  x ^= x >> 16; x *= 0x7feb352du; x ^= x >> 15; x *= 0x846ca68bu; x ^= x >> 16;
  return (float)((x >> 8) + 1u) * (1.0f / 16777216.0f);
}

// descending bitonic sort of SORTN (key, idx) pairs in LDS, 256 threads
__device__ __forceinline__ void bitonic_desc(float* key, int* idx, int tid) {
  for (unsigned k = 2; k <= SORTN; k <<= 1) {
    for (unsigned j = k >> 1; j > 0; j >>= 1) {
      __syncthreads();
      for (unsigned i = tid; i < SORTN; i += 256) {
        unsigned p = i ^ j;
        if (p > i) {
          bool dir = ((i & k) == 0);         // descending segment
          float a = key[i], c = key[p];
          if ((a < c) == dir) {
            key[i] = c; key[p] = a;
            int t = idx[i]; idx[i] = idx[p]; idx[p] = t;
          }
        }
      }
    }
  }
  __syncthreads();
}

// ---------------- Kernel A: IoU + sampling + deltas (1 block / image) -------
__global__ __launch_bounds__(256) void dtl_sample_kernel(
    const float* __restrict__ proposals,   // [B,2000,4]
    const int*   __restrict__ gtcls,       // [B,100]
    const float* __restrict__ gtbox,       // [B,100,4]
    const float* __restrict__ stddev,      // [4]
    float* __restrict__ out,
    float* __restrict__ ws) {
  const int b   = blockIdx.x;
  const int tid = threadIdx.x;

  __shared__ float sg[MAXGT][4];
  __shared__ float sga[MAXGT];
  __shared__ int   sgc[MAXGT];
  __shared__ int   sgv[MAXGT];
  __shared__ float ioumax[NPROP];
  __shared__ int   argm[NPROP];
  __shared__ unsigned char vprop[NPROP];
  __shared__ float skey[SORTN];
  __shared__ int   sidx[SORTN];
  __shared__ int   posIdx[NUMPOS];
  __shared__ float posKey[NUMPOS];
  __shared__ int   negNeed;

  // cache GT boxes / validity / areas
  for (int g = tid; g < MAXGT; g += 256) {
    float4 gb = ((const float4*)gtbox)[b * MAXGT + g];
    sg[g][0] = gb.x; sg[g][1] = gb.y; sg[g][2] = gb.z; sg[g][3] = gb.w;
    sga[g] = (gb.z - gb.x) * (gb.w - gb.y);
    int c = gtcls[b * MAXGT + g];
    sgc[g] = c; sgv[g] = (c > 0);
  }
  __syncthreads();

  // IoU max + argmax per proposal
  for (int i = tid; i < NPROP; i += 256) {
    float4 p = ((const float4*)proposals)[b * NPROP + i];
    vprop[i] = (p.x != 0.f) | (p.y != 0.f) | (p.z != 0.f) | (p.w != 0.f);
    float a1 = (p.z - p.x) * (p.w - p.y);
    float best = -2.f; int bi = 0;
    for (int g = 0; g < MAXGT; ++g) {
      float yA = fmaxf(p.x, sg[g][0]), xA = fmaxf(p.y, sg[g][1]);
      float yB = fminf(p.z, sg[g][2]), xB = fminf(p.w, sg[g][3]);
      float inter = fmaxf(yB - yA, 0.f) * fmaxf(xB - xA, 0.f);
      float uni   = fmaxf(a1 + sga[g] - inter, 1e-10f);
      float ov = sgv[g] ? (inter / uni) : -1.0f;
      if (ov > best) { best = ov; bi = g; }   // first-max like jnp.argmax
    }
    ioumax[i] = best; argm[i] = bi;
  }
  __syncthreads();

  // pass 1: random-score top-k over positives
  for (int i = tid; i < SORTN; i += 256) {
    float k = -2.f;
    if (i < NPROP) {
      bool pos = (ioumax[i] >= 0.5f) && vprop[i];
      k = pos ? rnd01(2u * (unsigned)b, (unsigned)i) : -1.f;
    }
    skey[i] = k; sidx[i] = i;
  }
  bitonic_desc(skey, sidx, tid);
  if (tid < NUMPOS) { posIdx[tid] = sidx[tid]; posKey[tid] = skey[tid]; }
  __syncthreads();
  if (tid == 0) {
    int P = 0;
    for (int t = 0; t < NUMPOS; ++t) P += (posKey[t] > 0.f);
    int need = (int)((float)P / 0.33f) - P;
    negNeed = min(max(need, 0), NUMNEG);
  }

  // pass 2: random-score top-k over negatives
  __syncthreads();
  for (int i = tid; i < SORTN; i += 256) {
    float k = -2.f;
    if (i < NPROP) {
      bool neg = (ioumax[i] < 0.5f) && vprop[i];
      k = neg ? rnd01(2u * (unsigned)b + 1u, (unsigned)i) : -1.f;
    }
    skey[i] = k; sidx[i] = i;
  }
  bitonic_desc(skey, sidx, tid);

  // emit rois / class_ids / deltas + workspace for the mask kernel
  const float s0 = stddev[0], s1 = stddev[1], s2 = stddev[2], s3 = stddev[3];
  float* wsb = ws + b * 400;   // [0,264): pos boxes, [264,330): valid, [330,396): gt idx
  if (tid < TRAIN) {
    const int obase = b * TRAIN + tid;
    if (tid < NUMPOS) {
      int   i   = posIdx[tid];
      float vld = (posKey[tid] > 0.f) ? 1.f : 0.f;
      float4 p  = ((const float4*)proposals)[b * NPROP + i];
      int   g   = argm[i];
      float gb0, gb1, gb2, gb3;
      if (vld > 0.f) { gb0 = sg[g][0]; gb1 = sg[g][1]; gb2 = sg[g][2]; gb3 = sg[g][3]; }
      else           { gb0 = p.x; gb1 = p.y; gb2 = p.z; gb3 = p.w; }
      float dh = p.z - p.x, dw = p.w - p.y;
      float cy = p.x + 0.5f * dh, cx = p.y + 0.5f * dw;
      float gh = gb2 - gb0, gw = gb3 - gb1;
      float gcy = gb0 + 0.5f * gh, gcx = gb1 + 0.5f * gw;
      float ih = 1.f / fmaxf(dh, 1e-10f), iw = 1.f / fmaxf(dw, 1e-10f);
      float d0 = (gcy - cy) * ih / s0;
      float d1 = (gcx - cx) * iw / s1;
      float d2 = __logf(fmaxf(gh * ih, 1e-10f)) / s2;
      float d3 = __logf(fmaxf(gw * iw, 1e-10f)) / s3;
      out[OFF_ROIS + obase * 4 + 0] = p.x * vld;
      out[OFF_ROIS + obase * 4 + 1] = p.y * vld;
      out[OFF_ROIS + obase * 4 + 2] = p.z * vld;
      out[OFF_ROIS + obase * 4 + 3] = p.w * vld;
      out[OFF_CLS  + obase] = (vld > 0.f) ? (float)sgc[g] : 0.f;
      out[OFF_DELT + obase * 4 + 0] = d0 * vld;
      out[OFF_DELT + obase * 4 + 1] = d1 * vld;
      out[OFF_DELT + obase * 4 + 2] = d2 * vld;
      out[OFF_DELT + obase * 4 + 3] = d3 * vld;
      wsb[tid * 4 + 0] = p.x; wsb[tid * 4 + 1] = p.y;
      wsb[tid * 4 + 2] = p.z; wsb[tid * 4 + 3] = p.w;
      wsb[264 + tid] = vld;
      wsb[330 + tid] = (float)g;
    } else {
      int   ns  = tid - NUMPOS;
      int   i   = sidx[ns];
      float vld = (skey[ns] > 0.f && ns < negNeed) ? 1.f : 0.f;
      float4 p  = ((const float4*)proposals)[b * NPROP + i];
      out[OFF_ROIS + obase * 4 + 0] = p.x * vld;
      out[OFF_ROIS + obase * 4 + 1] = p.y * vld;
      out[OFF_ROIS + obase * 4 + 2] = p.z * vld;
      out[OFF_ROIS + obase * 4 + 3] = p.w * vld;
      out[OFF_CLS  + obase] = 0.f;
      out[OFF_DELT + obase * 4 + 0] = 0.f;
      out[OFF_DELT + obase * 4 + 1] = 0.f;
      out[OFF_DELT + obase * 4 + 2] = 0.f;
      out[OFF_DELT + obase * 4 + 3] = 0.f;
    }
  }
}

// -------- Kernel C: mask crop_and_resize via WMMA (1 block / ROI slot) -----
// crop = Ry*M*Rx; row dim collapsed by gather-lerp (G1 = Ry*M, only 28 rows),
// column dim as dense f16 GEMM: out[32,32] = G1[32,160] x Rx[160,32] via
// 2x2 tiles of v_wmma_f32_16x16x32_f16, K = 5 chunks of 32.
__global__ __launch_bounds__(128) void dtl_mask_kernel(
    const float* __restrict__ masks,   // [B,160,160,100]
    float* __restrict__ out,
    const float* __restrict__ ws) {
  const int blk  = blockIdx.x;
  const int b    = blk / TRAIN;
  const int slot = blk - b * TRAIN;
  const int tid  = threadIdx.x;
  float* mo = out + OFF_MASK + (size_t)(b * TRAIN + slot) * (MH * MWD);
  const float* wsb = ws + b * 400;

  bool compute = false;
  float bx0 = 0.f, bx1 = 0.f, bx2 = 0.f, bx3 = 0.f;
  int g = 0;
  if (slot < NUMPOS) {
    float vld = wsb[264 + slot];            // block-uniform
    if (vld > 0.5f) {
      compute = true;
      bx0 = wsb[slot * 4 + 0]; bx1 = wsb[slot * 4 + 1];
      bx2 = wsb[slot * 4 + 2]; bx3 = wsb[slot * 4 + 3];
      g   = (int)wsb[330 + slot];
    }
  }
  if (!compute) {                           // zero-fill neg / invalid slots
    for (int i = tid; i < MH * MWD; i += 128) mo[i] = 0.f;
    return;
  }

  __shared__ __align__(16) _Float16 G1[32 * 160];  // A operand (row-lerped mask)
  __shared__ __align__(16) _Float16 BT[32 * 160];  // B operand, stored transposed
  __shared__ int   y0a[MH], y1a[MH], x0a[MH], x1a[MH];
  __shared__ float fya[MH], fxa[MH];

  if (tid < MH) {
    float t  = (float)tid * (1.0f / 27.0f);
    float ys = (bx0 + t * (bx2 - bx0)) * 159.0f;
    float xs = (bx1 + t * (bx3 - bx1)) * 159.0f;
    float yf = floorf(ys), xf = floorf(xs);
    fya[tid] = ys - yf;  fxa[tid] = xs - xf;
    int y0 = min(max((int)yf, 0), 159);
    int x0 = min(max((int)xf, 0), 159);
    y0a[tid] = y0; y1a[tid] = min(y0 + 1, 159);
    x0a[tid] = x0; x1a[tid] = min(x0 + 1, 159);
  }
  __syncthreads();

  // G1[i][x] = lerp over rows of mask channel g (strided global gathers)
  const float* mb = masks + (size_t)b * HH * WWD * MAXGT + g;
  __builtin_prefetch(mb, 0, 1);
  for (int idx = tid; idx < 32 * 160; idx += 128) {
    int i = idx / 160, x = idx - i * 160;
    _Float16 v = (_Float16)0.f;
    if (i < MH) {
      float fy = fya[i];
      const float* r0 = &mb[(y0a[i] * WWD + x) * MAXGT];
      const float* r1 = &mb[(y1a[i] * WWD + x) * MAXGT];
      __builtin_prefetch(r0 + 128 * MAXGT, 0, 1);
      v = (_Float16)((1.f - fy) * (*r0) + fy * (*r1));
    }
    G1[idx] = v;
    BT[idx] = (_Float16)0.f;
  }
  __syncthreads();
  if (tid < MH) {                           // Rx^T: 2 nonzeros per output col
    float fx = fxa[tid];
    int x0 = x0a[tid], x1 = x1a[tid];
    if (x1 == x0) BT[tid * 160 + x0] = (_Float16)1.0f;
    else { BT[tid * 160 + x0] = (_Float16)(1.f - fx); BT[tid * 160 + x1] = (_Float16)fx; }
  }
  __syncthreads();

  // 4 waves -> 4 output tiles (Mi,Ni in {0,1}); EXEC all-ones at WMMA site.
  const int wave = tid >> 5, lane = tid & 31;
  const int Mi = wave >> 1, Ni = wave & 1;
  const int sel = (lane >> 4) & 1;          // lane half selects K sub-block
  const int l15 = lane & 15;
  const int arow = Mi * 16 + l15;
  const int brow = Ni * 16 + l15;
  v8f_ acc = {0.f, 0.f, 0.f, 0.f, 0.f, 0.f, 0.f, 0.f};
#pragma unroll
  for (int kc = 0; kc < 5; ++kc) {
    const int base = kc * 32 + sel * 8;
    h8 a0 = *(const h8*)&G1[arow * 160 + base];       // K 0-7 / 8-15
    h8 a1 = *(const h8*)&G1[arow * 160 + base + 16];  // K 16-23 / 24-31
    h8 b0 = *(const h8*)&BT[brow * 160 + base];
    h8 b1 = *(const h8*)&BT[brow * 160 + base + 16];
    h16 A, Bf;
#pragma unroll
    for (int u = 0; u < 8; ++u) {
      A[u] = a0[u]; A[u + 8] = a1[u];
      Bf[u] = b0[u]; Bf[u + 8] = b1[u];
    }
    acc = __builtin_amdgcn_wmma_f32_16x16x32_f16(
        false, A, false, Bf, (short)0, acc, false, false);
  }
#pragma unroll
  for (int r = 0; r < 8; ++r) {             // C/D layout: VGPR r -> M=r (+8 hi lanes)
    int m = Mi * 16 + r + sel * 8;
    int n = Ni * 16 + l15;
    if (m < MH && n < MWD) mo[m * MWD + n] = rintf(acc[r]);
  }
}

extern "C" void kernel_launch(void* const* d_in, const int* in_sizes, int n_in,
                              void* d_out, int out_size, void* d_ws, size_t ws_size,
                              hipStream_t stream) {
  (void)in_sizes; (void)n_in; (void)out_size; (void)ws_size;
  const float* proposals = (const float*)d_in[0];
  const int*   gtcls     = (const int*)d_in[1];
  const float* gtbox     = (const float*)d_in[2];
  const float* gmasks    = (const float*)d_in[3];
  const float* stddev    = (const float*)d_in[4];
  float* out = (float*)d_out;
  float* ws  = (float*)d_ws;

  dtl_sample_kernel<<<BB, 256, 0, stream>>>(proposals, gtcls, gtbox, stddev, out, ws);
  dtl_mask_kernel<<<BB * TRAIN, 128, 0, stream>>>(gmasks, out, ws);
}
